// CTA_28381143891994
// MI455X (gfx1250) — compile-verified, matrix operand order checked by reading, not run on hardware
//
#include <hip/hip_runtime.h>
#include <hip/hip_bf16.h>
#include <math.h>

// ---------------------------------------------------------------------------
// MI455X (gfx1250) implementation.
//  - wave32, WMMA 16x16x32 bf16, weight-stationary recurrent kernel.
//  - 4 batch-groups x 16 WGs; gru_R B-fragments pinned in VGPRs (reused 2047x).
//  - One cluster/atomic barrier per step (z_new exchange via L2).
//  - Wk folded into wkq=Wk@q ; Wv deferred via Zacc = sum(alpha_t * z_t).
// ---------------------------------------------------------------------------

typedef __bf16 bf16_t;
typedef __attribute__((ext_vector_type(16))) __bf16 v16bf;
typedef __attribute__((ext_vector_type(8)))  __bf16 v8bf;
typedef __attribute__((ext_vector_type(8)))  float  v8f;

#define Bdim   64
#define Sdim   2048
#define Hdim   512
#define NSTEP  2047
#define GROUPS 4
#define WGS    16      // workgroups per batch-group
#define TREC   192     // 6 waves of 32

// ---- fragment index maps (per CDNA5 ISA 7.12.2, 16-bit A 16x32) -----------
__device__ __forceinline__ int kmap(int lane, int j) {
  int h = lane >> 4;
  return (j < 8) ? (h * 8 + j) : (16 + h * 8 + (j - 8));
}
// inverse: (kk in 0..31, row) -> (lane, j)
__device__ __forceinline__ void invk32(int kk, int r, int& ln, int& j) {
  if (kk < 16) { ln = r + ((kk >> 3) << 4); j = kk & 7; }
  else         { int k2 = kk - 16; ln = r + ((k2 >> 3) << 4); j = 8 + (k2 & 7); }
}
__device__ __forceinline__ float  bf2f(bf16_t x) { return (float)x; }
__device__ __forceinline__ bf16_t f2bf(float x)  { return (bf16_t)x; }
__device__ __forceinline__ float  sigmoidf_(float x) { return 1.f / (1.f + expf(-x)); }
__device__ __forceinline__ float  softplusf_(float x) { return (x > 20.f) ? x : log1pf(expf(x)); }

__device__ __forceinline__ v8f wmma_bf16(v16bf a, v16bf b, v8f c) {
  return __builtin_amdgcn_wmma_f32_16x16x32_bf16(false, a, false, b, (short)0, c, false, false);
}

// ---------------------------------------------------------------------------
// Pack an f32 KxN (row-major) matrix into WMMA B-fragment order:
//   dst[tile][chunk][lane][j], tile = col/16, chunk = k/32.
// ---------------------------------------------------------------------------
__global__ void pack_b_kernel(const float* __restrict__ src, bf16_t* __restrict__ dst,
                              int K, int N) {
  int chunks = K >> 5;
  long total = (long)(N >> 4) * chunks * 32 * 16;
  for (long idx = (long)blockIdx.x * blockDim.x + threadIdx.x; idx < total;
       idx += (long)gridDim.x * blockDim.x) {
    int j = idx & 15, lane = (idx >> 4) & 31;
    long rest = idx >> 9;
    int ch = (int)(rest % chunks), T = (int)(rest / chunks);
    int k = ch * 32 + kmap(lane, j);
    int col = T * 16 + (lane & 15);
    dst[idx] = f2bf(src[(long)k * N + col]);
  }
}

// ---------------------------------------------------------------------------
// qvec = query@Wq + bq ; wkq = Wk @ qvec ; qbk = qvec . bk    (runs once)
// ---------------------------------------------------------------------------
__global__ __launch_bounds__(256) void qk_kernel(
    const float* __restrict__ Wq, const float* __restrict__ bq,
    const float* __restrict__ query, const float* __restrict__ Wk,
    const float* __restrict__ bk, float* __restrict__ wkq, float* __restrict__ qbk) {
  __shared__ float qv[Hdim];
  int tid = threadIdx.x;
  for (int h = tid; h < Hdim; h += 256) {
    float s = bq[h];
    for (int i = 0; i < Hdim; ++i) s += query[i] * Wq[i * Hdim + h];
    qv[h] = s;
  }
  __syncthreads();
  for (int i = tid; i < Hdim; i += 256) {
    float s = 0.f;
    for (int j = 0; j < Hdim; ++j) s += Wk[i * Hdim + j] * qv[j];
    wkq[i] = s;
  }
  if (tid == 0) {
    float s = 0.f;
    for (int j = 0; j < Hdim; ++j) s += qv[j] * bk[j];
    qbk[0] = s;
  }
}

// ---------------------------------------------------------------------------
// proj = x @ Wp + bp  (bf16 out).  One 16x16 tile per wave, K=512.
// ---------------------------------------------------------------------------
__global__ __launch_bounds__(256) void gemm_proj_kernel(
    const float* __restrict__ x, const bf16_t* __restrict__ Wppack,
    const float* __restrict__ bp, bf16_t* __restrict__ projb) {
  int wid = (blockIdx.x * 256 + threadIdx.x) >> 5;
  int lane = threadIdx.x & 31;
  int rt = wid >> 5;          // 8192 row tiles (64*2048/16)
  int ct = wid & 31;          // 32 col tiles
  if (rt >= (Bdim * Sdim / 16)) return;
  int h = lane >> 4, rr = lane & 15;
  const float* ap = x + ((long)rt * 16 + rr) * Hdim;
  float bias = bp[ct * 16 + rr];
  v8f acc;
#pragma unroll
  for (int i = 0; i < 8; ++i) acc[i] = bias;
#pragma unroll
  for (int ch = 0; ch < 16; ++ch) {
    const float* p0 = ap + ch * 32 + h * 8;
    float4 f0 = *(const float4*)(p0);
    float4 f1 = *(const float4*)(p0 + 4);
    float4 f2 = *(const float4*)(p0 + 16);
    float4 f3 = *(const float4*)(p0 + 20);
    v16bf a;
    a[0]=f2bf(f0.x); a[1]=f2bf(f0.y); a[2]=f2bf(f0.z); a[3]=f2bf(f0.w);
    a[4]=f2bf(f1.x); a[5]=f2bf(f1.y); a[6]=f2bf(f1.z); a[7]=f2bf(f1.w);
    a[8]=f2bf(f2.x); a[9]=f2bf(f2.y); a[10]=f2bf(f2.z); a[11]=f2bf(f2.w);
    a[12]=f2bf(f3.x); a[13]=f2bf(f3.y); a[14]=f2bf(f3.z); a[15]=f2bf(f3.w);
    v16bf b = *(const v16bf*)(Wppack + (((long)ct * 16 + ch) * 32 + lane) * 16);
    acc = wmma_bf16(a, b, acc);
  }
  int col = ct * 16 + rr;
#pragma unroll
  for (int i = 0; i < 8; ++i) {
    int m = i + 8 * h;
    projb[((long)rt * 16 + m) * Hdim + col] = f2bf(acc[i]);
  }
}

// ---------------------------------------------------------------------------
// xg[b,t,:] = proj[b,t+1,:] @ gru_K + gru_b[0]   (bf16 out), t in [0,2047)
// ---------------------------------------------------------------------------
__global__ __launch_bounds__(256) void gemm_xg_kernel(
    const bf16_t* __restrict__ projb, const bf16_t* __restrict__ Kpack,
    const float* __restrict__ gru_b, bf16_t* __restrict__ xgout) {
  int wid = (blockIdx.x * 256 + threadIdx.x) >> 5;
  int lane = threadIdx.x & 31;
  const int NT = 96;                 // 1536/16 col tiles
  int rt = wid / NT;                 // 64*128 row tiles
  int ct = wid % NT;
  if (rt >= Bdim * 128) return;
  int b = rt >> 7, lt = rt & 127;
  int h = lane >> 4, rr = lane & 15;
  int tt = lt * 16 + rr;
  int ttc = (tt < NSTEP) ? tt : (NSTEP - 1);
  const bf16_t* ap = projb + (((long)b * Sdim) + (ttc + 1)) * Hdim;
  float bias = gru_b[ct * 16 + rr];
  v8f acc;
#pragma unroll
  for (int i = 0; i < 8; ++i) acc[i] = bias;
#pragma unroll
  for (int ch = 0; ch < 16; ++ch) {
    v8bf lo = *(const v8bf*)(ap + ch * 32 + h * 8);
    v8bf hi = *(const v8bf*)(ap + ch * 32 + 16 + h * 8);
    v16bf a = __builtin_shufflevector(lo, hi, 0,1,2,3,4,5,6,7,8,9,10,11,12,13,14,15);
    v16bf bfr = *(const v16bf*)(Kpack + (((long)ct * 16 + ch) * 32 + lane) * 16);
    acc = wmma_bf16(a, bfr, acc);
  }
  int col = ct * 16 + rr;
#pragma unroll
  for (int i = 0; i < 8; ++i) {
    int m = i + 8 * h;
    int tm = lt * 16 + m;
    if (tm < NSTEP)
      xgout[(((long)b * NSTEP) + tm) * 1536 + col] = f2bf(acc[i]);
  }
}

// ---------------------------------------------------------------------------
// init: zbufA (A-fragment order) from proj[:,0]; zero a and barrier counters.
// ---------------------------------------------------------------------------
__global__ void init_state_kernel(const bf16_t* __restrict__ projb,
                                  bf16_t* __restrict__ zbufA,
                                  float* __restrict__ abuf,
                                  unsigned* __restrict__ barCnt) {
  int idx = blockIdx.x * blockDim.x + threadIdx.x;
  if (idx < GROUPS * 16 * 32 * 16) {
    int j = idx & 15, lane = (idx >> 4) & 31, ch = (idx >> 9) & 15, G = idx >> 13;
    int r = lane & 15;
    int k = ch * 32 + kmap(lane, j);
    zbufA[idx] = projb[((long)(G * 16 + r) * Sdim) * Hdim + k];
  }
  if (idx < GROUPS * 16) abuf[idx] = 0.f;
  if (idx < GROUPS) barCnt[idx] = 0u;
}

// ---------------------------------------------------------------------------
// Group barrier: cluster barrier path (one wave signals per ISA 6.6) with an
// atomic-counter fallback when not dispatched as a cluster (ClusterID==0).
// ---------------------------------------------------------------------------
__device__ __forceinline__ void group_barrier(unsigned* cnt, unsigned target) {
  __syncthreads();
  __threadfence();
  if (__builtin_amdgcn_cluster_id_x() != 0) {
    if ((threadIdx.x >> 5) == 0)
      asm volatile("s_barrier_signal -3" ::: "memory");
    asm volatile("s_barrier_wait -3" ::: "memory");
  } else {
    if (threadIdx.x == 0) {
      __hip_atomic_fetch_add(cnt, 1u, __ATOMIC_RELEASE, __HIP_MEMORY_SCOPE_AGENT);
      while (__hip_atomic_load(cnt, __ATOMIC_ACQUIRE, __HIP_MEMORY_SCOPE_AGENT) < target)
        __builtin_amdgcn_s_sleep(1);
    }
  }
  __syncthreads();
  __threadfence();
}

struct RecParams {
  const bf16_t *Rpack, *W1pack, *W3pack, *xg;
  const float *ode_b1, *ode_W2, *ode_b2, *ode_b3;
  const float *gru_b, *wkq, *qbk, *ts, *Wv, *bv;
  bf16_t* zbufA;
  float *ZaccBuf, *abuf, *out;
  unsigned* barCnt;
};

// ---------------------------------------------------------------------------
// Persistent recurrent kernel: 64 WGs (4 groups x 16), 192 threads each.
// NOTE: fallback barrier requires all 64 WGs co-resident (43KB LDS, fine).
// ---------------------------------------------------------------------------
__global__ __launch_bounds__(TREC, 1) void recurrent_kernel(RecParams P) {
  const int G = blockIdx.x >> 4;       // batch group
  const int g = blockIdx.x & 15;       // 32-col slice owner
  const int tid = threadIdx.x;
  const int wv = tid >> 5, lane = tid & 31;
  const int h = lane >> 4, rr = lane & 15;

  __shared__ __align__(32) bf16_t zA[16][32][16];    // z   A-fragments (16KB)
  __shared__ __align__(32) bf16_t zhA[16][32][16];   // z_hat A-fragments (16KB)
  __shared__ __align__(32) bf16_t h2A[32][16];       // h2  A-fragment (1 chunk)
  __shared__ float h1s[16][32];
  __shared__ float Yst[6][16][16];                   // gru_R tile outputs
  __shared__ float Zacc[16][32];                     // running alpha*z slice
  __shared__ float scoreAcc[16], alphaS[16], aAccS[16];

  const float dt = softplusf_(P.ts[0]);
  const float qbk = P.qbk[0];
  const float inv_sqrt_h = 0.044194173824159216f;    // 1/sqrt(512)
  unsigned* cnt = P.barCnt + G;
  const bf16_t* zg_global = P.zbufA + (long)G * 8192;

  // --- pin this wave's gru_R B-fragments in VGPRs (reused every step) ---
  int part = wv >> 1, sub = wv & 1;
  int colR = part * 512 + g * 32 + sub * 16;         // col in 1536
  int TR = colR >> 4;
  v16bf Bfrag[16];
#pragma unroll
  for (int ch = 0; ch < 16; ++ch)
    Bfrag[ch] = *(const v16bf*)(P.Rpack + (((long)TR * 16 + ch) * 32 + lane) * 16);
  // ode_W3 fragments (tiles wv, wv+6, ..., K=32 -> single chunk)
  v16bf W3f[6];
#pragma unroll
  for (int k = 0; k < 6; ++k) {
    int t3 = wv + 6 * k;
    if (t3 < 32)
      W3f[k] = *(const v16bf*)(P.W3pack + ((long)t3 * 32 + lane) * 16);
  }

  for (int i = tid; i < 16 * 32; i += TREC) Zacc[i >> 5][i & 31] = 0.f;
  if (tid < 16) { aAccS[tid] = 0.f; scoreAcc[tid] = 0.f; }
  { // load initial z fragments
    const uint4* src = (const uint4*)zg_global;
    uint4* dst = (uint4*)&zA[0][0][0];
    for (int i = tid; i < 1024; i += TREC) dst[i] = src[i];
  }
  __syncthreads();

  unsigned nbar = 0;
  for (int t = 0; t < NSTEP; ++t) {
    // ---- S1: h1 = tanh(z@W1+b1) on waves 0/1 ; alpha partial dots ; prefetch
    if (wv < 2) {
      v8f acc = {};
#pragma unroll
      for (int ch = 0; ch < 16; ++ch) {
        v16bf a = *(const v16bf*)&zA[ch][lane][0];
        v16bf b = *(const v16bf*)(P.W1pack + (((long)wv * 16 + ch) * 32 + lane) * 16);
        acc = wmma_bf16(a, b, acc);
      }
      int col = wv * 16 + rr;
      float b1v = P.ode_b1[col];
#pragma unroll
      for (int i = 0; i < 8; ++i) h1s[i + 8 * h][col] = tanhf(acc[i] + b1v);
    }
    { // score partials: 16 rows x 12 segments
      int r = tid & 15, seg = tid >> 4;
      float s = 0.f;
      for (int k = seg; k < Hdim; k += 12) {
        int ln, j; invk32(k & 31, r, ln, j);
        s += bf2f(zA[k >> 5][ln][j]) * P.wkq[k];
      }
      atomicAdd(&scoreAcc[r], s);
    }
    if (t + 1 < NSTEP) { // prefetch next step's xg cachelines
      for (int idx = tid; idx < 48; idx += TREC) {
        int r = idx / 3, p = idx % 3;
        __builtin_prefetch(P.xg + (((long)(G * 16 + r)) * NSTEP + (t + 1)) * 1536
                               + p * 512 + g * 32, 0, 1);
      }
    }
    __syncthreads();

    // ---- S2: alpha ; h2 = tanh(h1@W2+b2) into A-fragment order
    if (tid < 16)
      alphaS[tid] = sigmoidf_((scoreAcc[tid] + qbk) * inv_sqrt_h);
    for (int idx = tid; idx < 512; idx += TREC) {
      int r = idx >> 5, c = idx & 31;
      float s = P.ode_b2[c];
      for (int k = 0; k < 32; ++k) s += h1s[r][k] * P.ode_W2[k * 32 + c];
      int ln, j; invk32(c, r, ln, j);
      h2A[ln][j] = f2bf(tanhf(s));
    }
    __syncthreads();

    // ---- S3: z_dot = h2@W3+b3 (WMMA, K=32); z_hat = z + dt*z_dot -> zhA
    {
      v16bf a = *(const v16bf*)&h2A[lane][0];
#pragma unroll
      for (int k = 0; k < 6; ++k) {
        int t3 = wv + 6 * k;
        if (t3 < 32) {
          v8f acc = {};
          acc = wmma_bf16(a, W3f[k], acc);
          int col = t3 * 16 + rr;
          float b3v = P.ode_b3[col];
          int ch = col >> 5, kk = col & 31;
#pragma unroll
          for (int i = 0; i < 8; ++i) {
            int m = i + 8 * h;
            int ln, j; invk32(kk, m, ln, j);
            zhA[ch][ln][j] = f2bf(bf2f(zA[ch][ln][j]) + dt * (acc[i] + b3v));
          }
        }
      }
    }
    // Zacc slice += alpha*z ; a accumulation ; re-zero score
    for (int idx = tid; idx < 512; idx += TREC) {
      int r = idx >> 5, cl = idx & 31;
      int kcol = g * 32 + cl;
      int ln, j; invk32(kcol & 31, r, ln, j);
      Zacc[r][cl] += alphaS[r] * bf2f(zA[kcol >> 5][ln][j]);
    }
    if (g == 0 && tid < 16) aAccS[tid] += dt * alphaS[tid];
    if (tid < 16) scoreAcc[tid] = 0.f;
    __syncthreads();

    // ---- S5: hg tile = z_hat @ gru_R(tile)  (weight-stationary WMMA)
    {
      v8f acc = {};
#pragma unroll
      for (int ch = 0; ch < 16; ++ch) {
        v16bf a = *(const v16bf*)&zhA[ch][lane][0];
        acc = wmma_bf16(a, Bfrag[ch], acc);
      }
#pragma unroll
      for (int i = 0; i < 8; ++i) Yst[wv][i + 8 * h][rr] = acc[i];
    }
    __syncthreads();

    // ---- S6: GRU gates (fully WG-local) -> z_new slice to global
    for (int idx = tid; idx < 512; idx += TREC) {
      int r = idx >> 5, cl = idx & 31;
      int colz = g * 32 + cl;
      float hz = Yst[0 + (cl >> 4)][r][cl & 15] + P.gru_b[1536 + colz];
      float hr = Yst[2 + (cl >> 4)][r][cl & 15] + P.gru_b[1536 + 512 + colz];
      float hh = Yst[4 + (cl >> 4)][r][cl & 15] + P.gru_b[1536 + 1024 + colz];
      long xbase = (((long)(G * 16 + r)) * NSTEP + t) * 1536;
      float xz = bf2f(P.xg[xbase + colz]);
      float xr = bf2f(P.xg[xbase + 512 + colz]);
      float xh = bf2f(P.xg[xbase + 1024 + colz]);
      int ln, j; invk32(colz & 31, r, ln, j);
      float zh = bf2f(zhA[colz >> 5][ln][j]);
      float zgate = sigmoidf_(xz + hz);
      float rgate = sigmoidf_(xr + hr);
      float cand = tanhf(xh + rgate * hh);
      float zn = zgate * zh + (1.f - zgate) * cand;
      P.zbufA[(long)G * 8192 + (colz >> 5) * 512 + ln * 16 + j] = f2bf(zn);
    }
    group_barrier(cnt, (++nbar) * WGS);
    { // reload full z fragments for next step
      const uint4* src = (const uint4*)zg_global;
      uint4* dst = (uint4*)&zA[0][0][0];
      for (int i = tid; i < 1024; i += TREC) dst[i] = src[i];
    }
    __syncthreads();
  }

  // ---- tail: c = dt*Zacc@Wv + a*bv ; out = z_final + c/a ------------------
  for (int idx = tid; idx < 512; idx += TREC) {
    int r = idx >> 5, cl = idx & 31;
    P.ZaccBuf[((long)(G * 16 + r)) * Hdim + g * 32 + cl] = Zacc[r][cl];
  }
  if (g == 0 && tid < 16) P.abuf[G * 16 + tid] = aAccS[tid];
  group_barrier(cnt, (++nbar) * WGS);
  for (int idx = tid; idx < 512; idx += TREC) {
    int r = idx >> 5, cl = idx & 31;
    int hcol = g * 32 + cl;
    const float* zr = P.ZaccBuf + ((long)(G * 16 + r)) * Hdim;
    float acc = 0.f;
    for (int k = 0; k < Hdim; ++k) acc += zr[k] * P.Wv[(long)k * Hdim + hcol];
    float a = P.abuf[G * 16 + r];
    int ln, j; invk32(hcol & 31, r, ln, j);
    float zfin = bf2f(zA[hcol >> 5][ln][j]);
    float corr = (a == 0.f) ? 0.f : (dt * acc) / a + P.bv[hcol];
    P.out[((long)(G * 16 + r)) * Hdim + hcol] = zfin + corr;
  }
}

// ---------------------------------------------------------------------------
static inline size_t alignUp(size_t x, size_t a) { return (x + a - 1) / a * a; }

extern "C" void kernel_launch(void* const* d_in, const int* in_sizes, int n_in,
                              void* d_out, int out_size, void* d_ws, size_t ws_size,
                              hipStream_t stream) {
  const float* x      = (const float*)d_in[0];
  const float* Wp     = (const float*)d_in[1];
  const float* bp     = (const float*)d_in[2];
  const float* ode_W1 = (const float*)d_in[3];
  const float* ode_b1 = (const float*)d_in[4];
  const float* ode_W2 = (const float*)d_in[5];
  const float* ode_b2 = (const float*)d_in[6];
  const float* ode_W3 = (const float*)d_in[7];
  const float* ode_b3 = (const float*)d_in[8];
  const float* Wq     = (const float*)d_in[9];
  const float* bq     = (const float*)d_in[10];
  const float* Wk     = (const float*)d_in[11];
  const float* bk     = (const float*)d_in[12];
  const float* Wv     = (const float*)d_in[13];
  const float* bv     = (const float*)d_in[14];
  const float* query  = (const float*)d_in[15];
  const float* tscale = (const float*)d_in[16];
  const float* gru_K  = (const float*)d_in[17];
  const float* gru_R  = (const float*)d_in[18];
  const float* gru_b  = (const float*)d_in[19];

  // workspace carve-out (assumes ws_size >= ~540MB; all offsets 256B aligned)
  char* w = (char*)d_ws;
  size_t off = 0;
  auto alloc = [&](size_t bytes) -> void* {
    void* p = w + off; off = alignUp(off + bytes, 256); return p;
  };
  bf16_t* Wppack = (bf16_t*)alloc((size_t)32 * 16 * 32 * 16 * 2);
  bf16_t* Kpack  = (bf16_t*)alloc((size_t)96 * 16 * 32 * 16 * 2);
  bf16_t* Rpack  = (bf16_t*)alloc((size_t)96 * 16 * 32 * 16 * 2);
  bf16_t* W1pack = (bf16_t*)alloc((size_t)2 * 16 * 32 * 16 * 2);
  bf16_t* W3pack = (bf16_t*)alloc((size_t)32 * 1 * 32 * 16 * 2);
  float*  wkq    = (float*)alloc(Hdim * 4);
  float*  qbk    = (float*)alloc(256);
  bf16_t* zbufA  = (bf16_t*)alloc((size_t)GROUPS * 8192 * 2);
  float*  abuf   = (float*)alloc(256);
  unsigned* barCnt = (unsigned*)alloc(256);
  float*  ZaccBuf = (float*)alloc((size_t)GROUPS * 16 * Hdim * 4);
  bf16_t* projb  = (bf16_t*)alloc((size_t)Bdim * Sdim * Hdim * 2);
  bf16_t* xg     = (bf16_t*)alloc((size_t)Bdim * NSTEP * 1536 * 2);
  (void)ws_size; (void)n_in; (void)in_sizes; (void)out_size;

  // 1) pack weights to WMMA fragment order (bf16)
  pack_b_kernel<<<512, 256, 0, stream>>>(Wp,     Wppack, 512, 512);
  pack_b_kernel<<<512, 256, 0, stream>>>(gru_K,  Kpack,  512, 1536);
  pack_b_kernel<<<512, 256, 0, stream>>>(gru_R,  Rpack,  512, 1536);
  pack_b_kernel<<<64,  256, 0, stream>>>(ode_W1, W1pack, 512, 32);
  pack_b_kernel<<<64,  256, 0, stream>>>(ode_W3, W3pack, 32,  512);
  // 2) fold attention query: wkq = Wk@(query@Wq+bq), qbk
  qk_kernel<<<1, 256, 0, stream>>>(Wq, bq, query, Wk, bk, wkq, qbk);
  // 3) proj = x@Wp + bp (bf16)
  gemm_proj_kernel<<<32768, 256, 0, stream>>>(x, Wppack, bp, projb);
  // 4) init z0 = proj[:,0] (fragment order), zero a + barrier counters
  init_state_kernel<<<128, 256, 0, stream>>>(projb, zbufA, abuf, barCnt);
  // 5) xg = proj[:,1:]@gru_K + gru_b[0] (bf16)
  gemm_xg_kernel<<<98304, 256, 0, stream>>>(projb, Kpack, gru_b, xg);
  // 6) persistent recurrent scan (4 groups x 16 WGs, 1 barrier/step)
  RecParams P;
  P.Rpack = Rpack; P.W1pack = W1pack; P.W3pack = W3pack; P.xg = xg;
  P.ode_b1 = ode_b1; P.ode_W2 = ode_W2; P.ode_b2 = ode_b2; P.ode_b3 = ode_b3;
  P.gru_b = gru_b; P.wkq = wkq; P.qbk = qbk; P.ts = tscale; P.Wv = Wv; P.bv = bv;
  P.zbufA = zbufA; P.ZaccBuf = ZaccBuf; P.abuf = abuf; P.out = (float*)d_out;
  P.barCnt = barCnt;
  recurrent_kernel<<<GROUPS * WGS, TREC, 0, stream>>>(P);
}
